// FactorizedAttention_39685497815450
// MI455X (gfx1250) — matrix operand order repaired and checked by reading, not compiled, not run on hardware
//
#include <hip/hip_runtime.h>
#include <hip/hip_bf16.h>

// ---------------------------------------------------------------------------
// FactorizedAttention for MI455X (gfx1250, wave32, WMMA 16x16x32 bf16 + TDM)
//
// Pipeline (all bf16 WMMA with f32 accumulation):
//   0. cast fp32 -> bf16 (inputs, conv_w, wq/wk/wv/wo, proj_w)
//   1. conv1d(k=3, SAME) as implicit-im2col GEMM  [16384 x 12288] x [12288 x 3072]
//   2. per-head QKV projections (3 GEMMs K=1024, N=1024)
//   3. block-local causal attention, one WG per (block, head), WMMA S and PV
//   4. output projection GEMM (K=1024, N=1024)
//   5. final dense GEMM (K=1024, N=4096) -> fp32 d_out
//
// gfx1250 specifics:
//   - A tiles (GEMM) and Q/K tiles (attention) staged by the Tensor Data
//     Mover (tensor_load_to_lds + s_wait_tensorcnt): DMA into LDS with zero
//     VGPR traffic and no DScnt pollution.
//   - double-buffered LDS + register-held software pipeline: tile s+1's TDM
//     and global loads issue before tile s's WMMAs, stores land after.
//   - batched fragment loads feeding back-to-back v_wmma_f32_16x16x32_bf16.
// ---------------------------------------------------------------------------

typedef __attribute__((ext_vector_type(16))) __bf16 v16bf;
typedef __attribute__((ext_vector_type(8)))  float v8f;
typedef __attribute__((ext_vector_type(4)))  unsigned int v4u;
typedef __attribute__((ext_vector_type(8)))  int v8i;
typedef __attribute__((ext_vector_type(4)))  int v4i;

union FragU {
    v16bf v;
    unsigned int   u[8];
    unsigned short s[16];
};

__device__ __forceinline__ unsigned short f2bf(float f) {
    unsigned u = __float_as_uint(f);
    unsigned r = ((u >> 16) & 1u) + 0x7fffu;   // round-to-nearest-even
    return (unsigned short)((u + r) >> 16);
}

// ---------------------------------------------------------------------------
// TDM: DMA a 2D bf16 tile (tile_k contiguous elems x tile_rows rows, row
// stride row_stride elems) from global memory into LDS at lds_byte_off.
// D# bitfields per CDNA5 ISA ch.8.3/8.4: group0 {count=1, lds_addr,
// global_addr[56:0], type=2}; group1 {data_size=2B, tensor_dim=tile_dim
// (no OOB), tile dims, dim0 stride}. Groups 2/3 zero (<=2D tensor).
// ---------------------------------------------------------------------------
__device__ __forceinline__ void tdm_load_tile_2d(unsigned lds_byte_off,
                                                 const unsigned short* gptr,
                                                 int tile_k, int tile_rows,
                                                 long long row_stride) {
    unsigned long long ga = (unsigned long long)(size_t)gptr;
    v4u g0 = { 1u,                                     // count=1, is_restore=0
               lds_byte_off,                           // lds_addr
               (unsigned)(ga & 0xffffffffull),         // global_addr lo
               (unsigned)((ga >> 32) & 0x1ffffffull) | (2u << 30) };  // hi|type=2
    v8i g1;
    g1[0] = 0x10000;                                   // data_size=1 -> 2 bytes
    g1[1] = (int)(((unsigned)tile_k & 0xffffu) << 16); // tensor_dim0[15:0]
    g1[2] = (int)((((unsigned)tile_k >> 16) & 0xffffu) |
                  (((unsigned)tile_rows & 0xffffu) << 16)); // td0 hi | td1 lo
    g1[3] = (int)((((unsigned)tile_rows >> 16) & 0xffffu) |
                  (((unsigned)tile_k & 0xffffu) << 16));    // td1 hi | tile_dim0
    g1[4] = (int)((unsigned)tile_rows & 0xffffu);      // tile_dim1 (tile_dim2=0)
    g1[5] = (int)(row_stride & 0xffffffffll);          // dim0 stride lo
    g1[6] = (int)((row_stride >> 32) & 0xffffll);      // dim0 stride hi
    g1[7] = 0;
    v4i z4 = {0, 0, 0, 0};
#if defined(__clang_major__) && (__clang_major__ >= 23)
    v8i z8 = {0, 0, 0, 0, 0, 0, 0, 0};
    __builtin_amdgcn_tensor_load_to_lds(g0, g1, z4, z4, z8, 0);
#else
    __builtin_amdgcn_tensor_load_to_lds(g0, g1, z4, z4, 0);
#endif
}

// ---------------------------------------------------------------------------
// 0. fp32 -> bf16 cast (grid-stride)
// ---------------------------------------------------------------------------
__global__ void cast_f32_bf16(const float* __restrict__ src,
                              unsigned short* __restrict__ dst, size_t n) {
    size_t i = (size_t)blockIdx.x * blockDim.x + threadIdx.x;
    size_t stride = (size_t)gridDim.x * blockDim.x;
    for (; i < n; i += stride) dst[i] = f2bf(src[i]);
}

// ---------------------------------------------------------------------------
// Generic bf16 GEMM: C[M,N] = A[M,K] * B[K,N] + bias
//   tile 128x128 per workgroup (256 threads = 8 wave32), double-buffered LDS,
//   software-pipelined staging (regs for tile s+1 loaded before compute of s).
//   mode==0: A tile staged by TDM; mode==1: implicit im2col (conv k=3 SAME).
// ---------------------------------------------------------------------------
__global__ __launch_bounds__(256)
void gemm_bf16(const unsigned short* __restrict__ A, int lda,
               const unsigned short* __restrict__ B, int ldb,
               const float* __restrict__ bias,
               void* __restrict__ Cv, int ldc,
               int K, int mode, int out_f32) {
    __shared__ __align__(16) unsigned short sa[2][128 * 32];   // A tiles [m][k]
    __shared__ __align__(16) unsigned short sbt[2][128 * 32];  // B tiles transposed [n][k]

    const int tid = threadIdx.x;
    const int l   = tid & 31;
    const int w   = tid >> 5;
    const int n0  = blockIdx.x * 128;
    const int m0  = blockIdx.y * 128;

    const v8f vzero = {0.f, 0.f, 0.f, 0.f, 0.f, 0.f, 0.f, 0.f};
    v8f acc[8];
#pragma unroll
    for (int i = 0; i < 8; ++i) acc[i] = vzero;

    const int nsteps = K >> 5;

    // ---- helpers (inlined): load tile 'kt' into registers / issue TDM ----
    auto load_a_regs = [&](int k0, uint4* areg) {
        const int t = k0 >> 12;              // conv tap (tile lies in one tap)
        const int d0 = k0 & 4095;
        const int rowshift = t - 1;
#pragma unroll
        for (int cc = 0; cc < 2; ++cc) {
            int c = tid + cc * 256;
            int row = c >> 2, off = (c & 3) << 3;
            int gm   = m0 + row;
            int lpos = (gm & 8191) + rowshift;     // L = 8192
            if ((unsigned)lpos < 8192u) {
                areg[cc] = *reinterpret_cast<const uint4*>(
                    A + (size_t)(gm + rowshift) * lda + d0 + off);
            } else {
                areg[cc].x = areg[cc].y = areg[cc].z = areg[cc].w = 0u;
            }
        }
    };
    auto load_b_regs = [&](int k0, uint4* breg) {
#pragma unroll
        for (int cc = 0; cc < 2; ++cc) {
            int c = tid + cc * 256;
            int row = c >> 4, off = (c & 15) << 3;
            const unsigned short* gsrc = B + (size_t)(k0 + row) * ldb + n0 + off;
            breg[cc] = *reinterpret_cast<const uint4*>(gsrc);
            if (k0 + 64 < K)                                // prefetch distance 2
                __builtin_prefetch(gsrc + (size_t)64 * ldb, 0, 0);
        }
    };
    auto store_a = [&](int buf, const uint4* areg) {
#pragma unroll
        for (int cc = 0; cc < 2; ++cc) {
            int c = tid + cc * 256;
            int row = c >> 2, off = (c & 3) << 3;
            *reinterpret_cast<uint4*>(&sa[buf][row * 32 + off]) = areg[cc];
        }
    };
    auto store_bt = [&](int buf, const uint4* breg) {
#pragma unroll
        for (int cc = 0; cc < 2; ++cc) {
            int c = tid + cc * 256;
            int row = c >> 4, off = (c & 15) << 3;
            const unsigned short* pv = reinterpret_cast<const unsigned short*>(&breg[cc]);
#pragma unroll
            for (int j = 0; j < 8; ++j) sbt[buf][(off + j) * 32 + row] = pv[j];
        }
    };

    // ---- prologue: stage tile 0 into buffer 0 ----
    {
        uint4 breg[2];
        load_b_regs(0, breg);
        store_bt(0, breg);
        if (mode == 0) {
            if (w == 0)
                tdm_load_tile_2d((unsigned)(size_t)&sa[0][0],
                                 A + (size_t)m0 * lda, 32, 128, (long long)lda);
        } else {
            uint4 areg[2];
            load_a_regs(0, areg);
            store_a(0, areg);
        }
    }

    for (int s = 0; s < nsteps; ++s) {
        const int buf = s & 1;
        if (mode == 0 && w == 0) __builtin_amdgcn_s_wait_tensorcnt(0);
        __syncthreads();   // tile s staged; buffer buf^1 free (readers passed)

        // --- issue next tile's DMA + global loads before computing ---
        uint4 areg[2], breg[2];
        const bool more = (s + 1 < nsteps);
        if (more) {
            const int kn = (s + 1) << 5;
            if (mode == 0) {
                if (w == 0)
                    tdm_load_tile_2d((unsigned)(size_t)&sa[buf ^ 1][0],
                                     A + (size_t)m0 * lda + kn, 32, 128,
                                     (long long)lda);
            } else {
                load_a_regs(kn, areg);
            }
            load_b_regs(kn, breg);
        }

        // --- compute tile s: batch fragments, 8 back-to-back WMMAs ---
        const int m = l & 15, g = l >> 4;
        FragU a;
#pragma unroll
        for (int eo = 0; eo < 8; ++eo) {
            int e = eo * 2;
            int k = e + (e >= 8 ? 8 : 0) + g * 8;      // ISA A-frag layout
            a.u[eo] = *reinterpret_cast<const unsigned*>(
                &sa[buf][(w * 16 + m) * 32 + k]);
        }
        FragU bfr[8];
#pragma unroll
        for (int nt = 0; nt < 8; ++nt) {
#pragma unroll
            for (int eo = 0; eo < 8; ++eo) {
                int e = eo * 2;
                int k = e + (e >= 8 ? 8 : 0) + g * 8;  // ISA B-frag layout
                bfr[nt].u[eo] = *reinterpret_cast<const unsigned*>(
                    &sbt[buf][(nt * 16 + m) * 32 + k]);
            }
        }
#pragma unroll
        for (int nt = 0; nt < 8; ++nt) {
            acc[nt] = __builtin_amdgcn_wmma_f32_16x16x32_bf16(
                false, a.v, false, bfr[nt].v, (short)0, acc[nt], false, false);
        }

        // --- store next tile's registers into the alternate buffer ---
        if (more) {
            if (mode != 0) store_a(buf ^ 1, areg);
            store_bt(buf ^ 1, breg);
        }
    }

    // --- epilogue: bias add, store (C layout: comp r -> row r / 8+r) ---
    const int g = l >> 4, nl = l & 15;
#pragma unroll
    for (int nt = 0; nt < 8; ++nt) {
        int col = n0 + nt * 16 + nl;
        float bv = bias ? bias[col] : 0.f;
#pragma unroll
        for (int r = 0; r < 8; ++r) {
            int mrow = m0 + w * 16 + r + g * 8;
            float v = acc[nt][r] + bv;
            if (out_f32)
                reinterpret_cast<float*>(Cv)[(size_t)mrow * ldc + col] = v;
            else
                reinterpret_cast<unsigned short*>(Cv)[(size_t)mrow * ldc + col] = f2bf(v);
        }
    }
}

// ---------------------------------------------------------------------------
// Block-local causal attention: one WG per (block bb, head h).
//   Q,K staged by TDM (row-major 128x64, stride 1024); V transposed manually.
//   S = Q K^T (WMMA), scale, causal mask, softmax (wave-local shfl butterflies)
//   O = P V (WMMA) -> oh bf16
// LDS: q 16K | k 16K | vT 16K ; P (32K) reuses q+k region. Total 48 KB.
// ---------------------------------------------------------------------------
__global__ __launch_bounds__(256)
void block_attention(const unsigned short* __restrict__ qh,
                     const unsigned short* __restrict__ kh,
                     const unsigned short* __restrict__ vh,
                     unsigned short* __restrict__ oh) {
    __shared__ __align__(16) unsigned short smem[24576];   // 48 KB
    unsigned short* sq = smem;            // [128][64]
    unsigned short* sk = smem + 8192;     // [128][64] (row-major == B^T for S)
    unsigned short* sv = smem + 16384;    // V transposed: [64 e][128 key]
    unsigned short* sp = smem;            // P [128][128], reuses q+k

    const int tid = threadIdx.x;
    const int l   = tid & 31;
    const int w   = tid >> 5;
    const int bb  = blockIdx.x >> 4;
    const int h   = blockIdx.x & 15;
    const size_t rowbase = (size_t)bb * 128;
    const int    colbase = h * 64;

    // --- stage: Q,K via TDM; V transposed by threads ---
    if (w == 0) {
        tdm_load_tile_2d((unsigned)(size_t)sq,
                         qh + rowbase * 1024 + colbase, 64, 128, 1024);
        tdm_load_tile_2d((unsigned)(size_t)sk,
                         kh + rowbase * 1024 + colbase, 64, 128, 1024);
    }
#pragma unroll
    for (int cc = 0; cc < 4; ++cc) {
        int c = tid + cc * 256;
        int row = c >> 3, off = (c & 7) << 3;
        uint4 vv = *reinterpret_cast<const uint4*>(
            vh + (rowbase + row) * 1024 + colbase + off);
        const unsigned short* pv = reinterpret_cast<const unsigned short*>(&vv);
#pragma unroll
        for (int j = 0; j < 8; ++j) sv[(off + j) * 128 + row] = pv[j];
    }
    if (w == 0) __builtin_amdgcn_s_wait_tensorcnt(0);
    __syncthreads();

    const v8f vzero = {0.f, 0.f, 0.f, 0.f, 0.f, 0.f, 0.f, 0.f};
    const int m = l & 15, g = l >> 4;

    // --- S = Q K^T : wave w -> rows [w*16, w*16+16), 8 key tiles, K=64 ---
    v8f acc[8];
#pragma unroll
    for (int i = 0; i < 8; ++i) acc[i] = vzero;
#pragma unroll
    for (int kt = 0; kt < 2; ++kt) {
        FragU a;
#pragma unroll
        for (int eo = 0; eo < 8; ++eo) {
            int e = eo * 2;
            int k = e + (e >= 8 ? 8 : 0) + g * 8 + kt * 32;
            a.u[eo] = *reinterpret_cast<const unsigned*>(sq + (w * 16 + m) * 64 + k);
        }
        FragU bfr[8];
#pragma unroll
        for (int nk = 0; nk < 8; ++nk) {
#pragma unroll
            for (int eo = 0; eo < 8; ++eo) {
                int e = eo * 2;
                int k = e + (e >= 8 ? 8 : 0) + g * 8 + kt * 32;
                bfr[nk].u[eo] =
                    *reinterpret_cast<const unsigned*>(sk + (nk * 16 + m) * 64 + k);
            }
        }
#pragma unroll
        for (int nk = 0; nk < 8; ++nk) {
            acc[nk] = __builtin_amdgcn_wmma_f32_16x16x32_bf16(
                false, a.v, false, bfr[nk].v, (short)0, acc[nk], false, false);
        }
    }

    // --- scale + causal mask + softmax (rows live in 16-lane groups) ---
    const float scale = 0.125f;   // 1/sqrt(64)
    float mx[8], sum[8];
#pragma unroll
    for (int r = 0; r < 8; ++r) mx[r] = -1e30f;
#pragma unroll
    for (int nk = 0; nk < 8; ++nk) {
#pragma unroll
        for (int r = 0; r < 8; ++r) {
            int mrow = w * 16 + r + g * 8;
            int ncol = nk * 16 + m;
            float s = acc[nk][r] * scale;
            if (ncol > mrow) s = -1e9f;
            acc[nk][r] = s;
            mx[r] = fmaxf(mx[r], s);
        }
    }
#pragma unroll
    for (int o = 8; o >= 1; o >>= 1)
#pragma unroll
        for (int r = 0; r < 8; ++r)
            mx[r] = fmaxf(mx[r], __shfl_xor(mx[r], o, 32));
#pragma unroll
    for (int r = 0; r < 8; ++r) sum[r] = 0.f;
#pragma unroll
    for (int nk = 0; nk < 8; ++nk)
#pragma unroll
        for (int r = 0; r < 8; ++r) {
            float e = __expf(acc[nk][r] - mx[r]);
            acc[nk][r] = e;
            sum[r] += e;
        }
#pragma unroll
    for (int o = 8; o >= 1; o >>= 1)
#pragma unroll
        for (int r = 0; r < 8; ++r)
            sum[r] += __shfl_xor(sum[r], o, 32);

    __syncthreads();   // all waves done reading Q/K before P overwrites them
#pragma unroll
    for (int nk = 0; nk < 8; ++nk)
#pragma unroll
        for (int r = 0; r < 8; ++r) {
            int mrow = w * 16 + r + g * 8;
            sp[mrow * 128 + nk * 16 + m] = f2bf(acc[nk][r] / sum[r]);
        }
    __syncthreads();

    // --- O = P V : rows w*16.., N=64 (4 tiles), K=128 (4 steps) ---
    v8f oacc[4];
#pragma unroll
    for (int i = 0; i < 4; ++i) oacc[i] = vzero;
#pragma unroll
    for (int kt = 0; kt < 4; ++kt) {
        FragU a;
#pragma unroll
        for (int eo = 0; eo < 8; ++eo) {
            int e = eo * 2;
            int k = e + (e >= 8 ? 8 : 0) + g * 8 + kt * 32;
            a.u[eo] = *reinterpret_cast<const unsigned*>(sp + (w * 16 + m) * 128 + k);
        }
        FragU bfr[4];
#pragma unroll
        for (int nv = 0; nv < 4; ++nv) {
#pragma unroll
            for (int eo = 0; eo < 8; ++eo) {
                int e = eo * 2;
                int k = e + (e >= 8 ? 8 : 0) + g * 8 + kt * 32;
                bfr[nv].u[eo] =
                    *reinterpret_cast<const unsigned*>(sv + (nv * 16 + m) * 128 + k);
            }
        }
#pragma unroll
        for (int nv = 0; nv < 4; ++nv) {
            oacc[nv] = __builtin_amdgcn_wmma_f32_16x16x32_bf16(
                false, a.v, false, bfr[nv].v, (short)0, oacc[nv], false, false);
        }
    }
#pragma unroll
    for (int nv = 0; nv < 4; ++nv)
#pragma unroll
        for (int r = 0; r < 8; ++r) {
            int mrow = w * 16 + r + g * 8;
            oh[(rowbase + mrow) * 1024 + colbase + nv * 16 + m] = f2bf(oacc[nv][r]);
        }
}

// ---------------------------------------------------------------------------
// Host launch
// ---------------------------------------------------------------------------
extern "C" void kernel_launch(void* const* d_in, const int* in_sizes, int n_in,
                              void* d_out, int out_size, void* d_ws, size_t ws_size,
                              hipStream_t stream) {
    (void)in_sizes; (void)n_in; (void)out_size; (void)ws_size;

    const float* inputs = (const float*)d_in[0];
    const float* conv_w = (const float*)d_in[1];
    const float* conv_b = (const float*)d_in[2];
    const float* wq     = (const float*)d_in[3];
    const float* bq     = (const float*)d_in[4];
    const float* wk     = (const float*)d_in[5];
    const float* bk     = (const float*)d_in[6];
    const float* wv     = (const float*)d_in[7];
    const float* bv     = (const float*)d_in[8];
    const float* wo     = (const float*)d_in[9];
    const float* bo     = (const float*)d_in[10];
    const float* proj_w = (const float*)d_in[11];
    const float* proj_b = (const float*)d_in[12];
    float* out = (float*)d_out;

    // workspace layout (bf16 raw ushort), 256B-aligned offsets
    char* ws = (char*)d_ws;
    unsigned short* x_bf  = (unsigned short*)(ws + 0);            // 16384x4096
    unsigned short* cw_bf = (unsigned short*)(ws + 134217728ull); // 12288x3072
    unsigned short* wq_bf = (unsigned short*)(ws + 209715200ull); // 1024x1024
    unsigned short* wk_bf = (unsigned short*)(ws + 211812352ull);
    unsigned short* wv_bf = (unsigned short*)(ws + 213909504ull);
    unsigned short* wo_bf = (unsigned short*)(ws + 216006656ull);
    unsigned short* pw_bf = (unsigned short*)(ws + 218103808ull); // 1024x4096
    unsigned short* qkv   = (unsigned short*)(ws + 226492416ull); // 16384x3072
    unsigned short* qh    = (unsigned short*)(ws + 327155712ull); // 16384x1024
    unsigned short* kh    = (unsigned short*)(ws + 360710144ull);
    unsigned short* vh    = (unsigned short*)(ws + 394264576ull);
    unsigned short* oh    = (unsigned short*)(ws + 427819008ull);
    unsigned short* ao    = (unsigned short*)(ws + 461373440ull);

    auto cast = [&](const float* s, unsigned short* d, size_t n) {
        int blocks = (int)((n + 1023) / 1024);
        if (blocks > 65535) blocks = 65535;
        cast_f32_bf16<<<blocks, 1024, 0, stream>>>(s, d, n);
    };
    cast(inputs, x_bf,  (size_t)16384 * 4096);
    cast(conv_w, cw_bf, (size_t)12288 * 3072);
    cast(wq,     wq_bf, (size_t)1024 * 1024);
    cast(wk,     wk_bf, (size_t)1024 * 1024);
    cast(wv,     wv_bf, (size_t)1024 * 1024);
    cast(wo,     wo_bf, (size_t)1024 * 1024);
    cast(proj_w, pw_bf, (size_t)1024 * 4096);

    // 1. conv-as-GEMM (implicit im2col): [16384 x 12288] x [12288 x 3072]
    gemm_bf16<<<dim3(24, 128), 256, 0, stream>>>(
        x_bf, 4096, cw_bf, 3072, conv_b, qkv, 3072, 12288, /*mode=*/1, /*f32=*/0);

    // 2. per-head QKV projections (q/k/v are column slices of qkv, lda=3072)
    gemm_bf16<<<dim3(8, 128), 256, 0, stream>>>(
        qkv + 0,    3072, wq_bf, 1024, bq, qh, 1024, 1024, 0, 0);
    gemm_bf16<<<dim3(8, 128), 256, 0, stream>>>(
        qkv + 1024, 3072, wk_bf, 1024, bk, kh, 1024, 1024, 0, 0);
    gemm_bf16<<<dim3(8, 128), 256, 0, stream>>>(
        qkv + 2048, 3072, wv_bf, 1024, bv, vh, 1024, 1024, 0, 0);

    // 3. block-local causal attention: 128 blocks x 16 heads
    block_attention<<<2048, 256, 0, stream>>>(qh, kh, vh, oh);

    // 4. MHA output projection: [16384x1024] x [1024x1024] + bo
    gemm_bf16<<<dim3(8, 128), 256, 0, stream>>>(
        oh, 1024, wo_bf, 1024, bo, ao, 1024, 1024, 0, 0);

    // 5. final dense -> fp32 out: [16384x1024] x [1024x4096] + proj_b
    gemm_bf16<<<dim3(32, 128), 256, 0, stream>>>(
        ao, 1024, pw_bf, 4096, proj_b, out, 4096, 1024, 0, /*f32=*/1);
}
